// Model_70755291234793
// MI455X (gfx1250) — matrix operand order repaired and checked by reading, not compiled, not run on hardware
//
#include <hip/hip_runtime.h>

// ---------------- model dims ----------------
#define Hh   512
#define Bb   64
#define TEe  128
#define TDd  512
#define Ee   256
#define KATT 10
#define NMIXm 20
#define SCALEK (1.0f/25.0f)

// ---------------- vector types ----------------
typedef __attribute__((ext_vector_type(16))) __bf16        bf16x16;
typedef __attribute__((ext_vector_type(8)))  float         f32x8;
typedef __attribute__((ext_vector_type(4)))  unsigned int  v4u;
typedef __attribute__((ext_vector_type(8)))  unsigned int  v8u;

// ---------------- output offsets (flat d_out, reference return order) ------
constexpr size_t O_MUS  = 0;
constexpr size_t O_SIG  = O_MUS  + (size_t)TDd*Bb*NMIXm*2;
constexpr size_t O_CORR = O_SIG  + (size_t)TDd*Bb*NMIXm*2;
constexpr size_t O_LC   = O_CORR + (size_t)TDd*Bb*NMIXm;
constexpr size_t O_BERN = O_LC   + (size_t)TDd*Bb*NMIXm;
constexpr size_t O_AH   = O_BERN + (size_t)TDd*Bb;
constexpr size_t O_AC   = O_AH   + (size_t)TDd*Bb*Hh;
constexpr size_t O_AK   = O_AC   + (size_t)TDd*Bb*Hh;
constexpr size_t O_AW   = O_AK   + (size_t)TDd*Bb*KATT;
constexpr size_t O_APHI = O_AW   + (size_t)TDd*Bb*Hh;
constexpr size_t O_H1   = O_APHI + (size_t)TDd*Bb*TEe;
constexpr size_t O_C1   = O_H1   + (size_t)TDd*Bb*Hh;
constexpr size_t O_H2   = O_C1   + (size_t)TDd*Bb*Hh;
constexpr size_t O_C2   = O_H2   + (size_t)TDd*Bb*Hh;

// ---------------- helpers ----------------
__device__ __forceinline__ unsigned short f2bf(float f) {
  unsigned int u = __float_as_uint(f);
  u += 0x7FFFu + ((u >> 16) & 1u);            // round-to-nearest-even
  return (unsigned short)(u >> 16);
}
__device__ __forceinline__ float sigm(float x){ return 1.0f/(1.0f+__expf(-x)); }

__device__ __forceinline__ f32x8 wmma_bf16(bf16x16 a, bf16x16 b, f32x8 c) {
  return __builtin_amdgcn_wmma_f32_16x16x32_bf16(false, a, false, b, (short)0, c, false, false);
}

// A fragment (16x32 bf16, row-major A with leading-dim lda).
// ISA layout: lanes 0-15 row M=lane, K = {kc..kc+7, kc+16..kc+23};
//             lanes 16-31 row M=lane-16, K = {kc+8..kc+15, kc+24..kc+31}.
__device__ __forceinline__ bf16x16 load_a_frag(const unsigned short* __restrict__ A,
                                               int lda, int row, int kc, int lane) {
  int klo = kc + ((lane & 16) ? 8 : 0);
  const unsigned short* p = A + (size_t)row * lda + klo;
  v4u a0 = *reinterpret_cast<const v4u*>(p);
  v4u a1 = *reinterpret_cast<const v4u*>(p + 16);
  v8u t = { a0.x, a0.y, a0.z, a0.w, a1.x, a1.y, a1.z, a1.w };
  return (bf16x16)t;
}
// B fragment (32x16 bf16) from N-major (transposed) weight Bt with leading-dim ldb.
// ISA layout: lanes 0-15 col N=lane, K = {kc..kc+7, kc+16..kc+23}; lanes 16-31 col N=lane-16, K = {kc+8..kc+15, kc+24..kc+31}.
__device__ __forceinline__ bf16x16 load_b_frag(const unsigned short* __restrict__ Bt,
                                               int ldb, int col, int kc, int lane) {
  int kb = kc + ((lane & 16) ? 8 : 0);
  const unsigned short* p = Bt + (size_t)col * ldb + kb;
  v4u a0 = *reinterpret_cast<const v4u*>(p);
  v4u a1 = *reinterpret_cast<const v4u*>(p + 16);
  v8u t = { a0.x, a0.y, a0.z, a0.w, a1.x, a1.y, a1.z, a1.w };
  return (bf16x16)t;
}

// ---------------- generic bf16 WMMA GEMM: C(MxN) = A(MxK) * Bt(NxK)^T [+bias][+=C] ---
// grid = (N/64, M/128), block = 128 (4 waves).
// Wave w owns rows [blockIdx.y*128 + 32w, +32) as 2 M-tiles x 4 N-tiles: each B
// fragment feeds two WMMAs -> 21.3 flops/byte of cache traffic (vs 12.8 at 1x4).
__global__ __launch_bounds__(128)
void gemm_bf16(const unsigned short* __restrict__ A, int lda,
               const unsigned short* __restrict__ Bt, int ldb,
               float* __restrict__ C, int ldc, int K,
               const float* __restrict__ bias, int accumulate) {
  int lane = threadIdx.x & 31;
  int w    = threadIdx.x >> 5;
  int m0   = blockIdx.y * 128 + w * 32;
  int n0   = blockIdx.x * 64;
  int cl   = lane & 15;
  f32x8 acc[2][4] = {{f32x8{}, f32x8{}, f32x8{}, f32x8{}},
                     {f32x8{}, f32x8{}, f32x8{}, f32x8{}}};
  for (int kc = 0; kc < K; kc += 32) {
    bf16x16 a0 = load_a_frag(A, lda, m0 + cl,      kc, lane);
    bf16x16 a1 = load_a_frag(A, lda, m0 + 16 + cl, kc, lane);
#pragma unroll
    for (int nt = 0; nt < 4; ++nt) {
      bf16x16 bf_ = load_b_frag(Bt, ldb, n0 + nt*16 + cl, kc, lane);
      acc[0][nt] = wmma_bf16(a0, bf_, acc[0][nt]);
      acc[1][nt] = wmma_bf16(a1, bf_, acc[1][nt]);
    }
  }
#pragma unroll
  for (int mt = 0; mt < 2; ++mt) {
    int rbase = m0 + mt*16 + ((lane & 16) ? 8 : 0);
#pragma unroll
    for (int nt = 0; nt < 4; ++nt) {
      int col = n0 + nt*16 + cl;
      float bv = bias ? bias[col] : 0.0f;
#pragma unroll
      for (int r = 0; r < 8; ++r) {
        size_t idx = (size_t)(rbase + r) * ldc + col;
        float v = acc[mt][nt][r] + bv;
        if (accumulate) v += C[idx];
        C[idx] = v;
      }
    }
  }
}

// ---------------- fused recurrent step: gates = Gpre_t + A0@Bt0 [+ A1@Bt1]; LSTM ----
// grid = H/16 = 32 blocks, block 128 (4 waves). Each block owns h-columns [16j,16j+16)
// and computes the 4 gate tiles (col offsets g*H) so the pointwise LSTM fuses in regs.
__global__ __launch_bounds__(128)
void lstm_step(const float* __restrict__ Gpre_t,
               const unsigned short* __restrict__ A0,
               const unsigned short* __restrict__ Bt0, int ldb0,
               const unsigned short* __restrict__ A1,
               const unsigned short* __restrict__ Bt1, int ldb1,
               const float* __restrict__ hprev, const float* __restrict__ cprev,
               const float* __restrict__ mask_t,
               float* __restrict__ hout, float* __restrict__ cout,
               unsigned short* __restrict__ hbfout) {
  int lane = threadIdx.x & 31;
  int w    = threadIdx.x >> 5;
  int j    = blockIdx.x * 16;
  int cl   = lane & 15;
  int rowA = w * 16 + cl;
  int rbase = w * 16 + ((lane & 16) ? 8 : 0);

  f32x8 acc[4];
#pragma unroll
  for (int g = 0; g < 4; ++g)
#pragma unroll
    for (int r = 0; r < 8; ++r)
      acc[g][r] = Gpre_t[(size_t)(rbase + r) * (4*Hh) + g*Hh + j + cl];

  for (int kc = 0; kc < Hh; kc += 32) {
    bf16x16 af = load_a_frag(A0, Hh, rowA, kc, lane);
#pragma unroll
    for (int g = 0; g < 4; ++g) {
      bf16x16 bf_ = load_b_frag(Bt0, ldb0, g*Hh + j + cl, kc, lane);
      acc[g] = wmma_bf16(af, bf_, acc[g]);
    }
  }
  if (A1) {
    for (int kc = 0; kc < Hh; kc += 32) {
      bf16x16 af = load_a_frag(A1, Hh, rowA, kc, lane);
#pragma unroll
      for (int g = 0; g < 4; ++g) {
        bf16x16 bf_ = load_b_frag(Bt1, ldb1, g*Hh + j + cl, kc, lane);
        acc[g] = wmma_bf16(af, bf_, acc[g]);
      }
    }
  }
#pragma unroll
  for (int r = 0; r < 8; ++r) {
    int row = rbase + r;
    size_t idx = (size_t)row * Hh + j + cl;
    float cp = cprev[idx], hp = hprev[idx], m = mask_t[row];
    float iv = sigm(acc[0][r]);
    float fv = sigm(acc[1][r]);
    float ov = sigm(acc[2][r]);
    float gv = tanhf(acc[3][r]);
    float cn = fv * cp + iv * gv;
    float hn = ov * tanhf(cn);
    float cf = m * cn + (1.0f - m) * cp;
    float hf = m * hn + (1.0f - m) * hp;
    cout[idx] = cf; hout[idx] = hf; hbfout[idx] = f2bf(hf);
  }
}

// ---------------- attention head: att = h@Watt^T + batt; k update; phi --------------
// single workgroup, 256 threads (8 waves -> 4x2 tiles of the 64x32 att matrix)
__global__ __launch_bounds__(256)
void attn_phi_kernel(const unsigned short* __restrict__ hbf,
                     const unsigned short* __restrict__ Wattt,
                     const float* __restrict__ batt,
                     const float* __restrict__ kprev,
                     const float* __restrict__ mask_t,
                     const float* __restrict__ encmask,
                     float* __restrict__ kout, float* __restrict__ phiout) {
  __shared__ float att[64][32];
  __shared__ float aL[64][KATT], bL[64][KATT], kL[64][KATT];
  int lane = threadIdx.x & 31;
  int w    = threadIdx.x >> 5;
  int mt = w & 3, nt = w >> 2;
  int cl = lane & 15;
  int rowA = mt*16 + cl;
  f32x8 acc = {};
  for (int kc = 0; kc < Hh; kc += 32) {
    bf16x16 af  = load_a_frag(hbf, Hh, rowA, kc, lane);
    bf16x16 bf_ = load_b_frag(Wattt, Hh, nt*16 + cl, kc, lane);
    acc = wmma_bf16(af, bf_, acc);
  }
  int rbase = mt*16 + ((lane & 16) ? 8 : 0);
  int col   = nt*16 + cl;
  float bv = (col < 3*KATT) ? batt[col] : 0.0f;
#pragma unroll
  for (int r = 0; r < 8; ++r) att[rbase + r][col] = acc[r] + bv;
  __syncthreads();

  for (int idx = threadIdx.x; idx < 64*KATT; idx += 256) {
    int b = idx / KATT, jj = idx % KATT;
    float m  = mask_t[b];
    float av = __expf(att[b][jj]);
    float bw = __expf(att[b][KATT + jj]);
    float kp = kprev[b*KATT + jj];
    float kn = kp + SCALEK * __expf(att[b][2*KATT + jj]);
    float kk = m*kn + (1.0f - m)*kp;
    kout[b*KATT + jj] = kk;
    aL[b][jj] = av; bL[b][jj] = bw; kL[b][jj] = kk;
  }
  __syncthreads();

  for (int idx = threadIdx.x; idx < 64*TEe; idx += 256) {
    int b = idx >> 7, u = idx & (TEe - 1);
    float uu = (float)u, s = 0.0f;
#pragma unroll
    for (int jj = 0; jj < KATT; ++jj) {
      float d = kL[b][jj] - uu;
      s += aL[b][jj] * __expf(-bL[b][jj]*d*d);
    }
    s *= encmask[u*Bb + b];
    phiout[b*TEe + u] = s;
  }
}

// ---------------- attention context: w[b,h] = sum_u phi[b,u]*p_e[u,b,h] -------------
__global__ __launch_bounds__(256)
void attn_w_kernel(const float* __restrict__ phi_t, const float* __restrict__ pe,
                   const float* __restrict__ wprev, const float* __restrict__ mask_t,
                   float* __restrict__ wout, unsigned short* __restrict__ wbfout) {
  int g = blockIdx.x * 256 + threadIdx.x;     // 32768 = B*H
  int b = g >> 9, h = g & (Hh - 1);
  const float* pp = pe + (size_t)b * Hh + h;
  const float* ph = phi_t + (size_t)b * TEe;
  float acc = 0.0f;
  for (int u = 0; u < TEe; ++u) acc += ph[u] * pp[(size_t)u * (Bb*Hh)];
  float m = mask_t[b];
  float wv = m * acc + (1.0f - m) * wprev[g];
  wout[g] = wv; wbfout[g] = f2bf(wv);
}

// ---------------- precompute kernels ----------------
__global__ __launch_bounds__(256)
void transpose_w_kernel(const float* __restrict__ W, unsigned short* __restrict__ Wt,
                        int K, int N, int Npad) {
  int idx = blockIdx.x * 256 + threadIdx.x;
  if (idx >= Npad * K) return;
  int n = idx / K, k = idx - n * K;
  float v = (n < N) ? W[(size_t)k * N + n] : 0.0f;
  Wt[idx] = f2bf(v);
}

__global__ __launch_bounds__(256)
void cvt_bf16_kernel(const float* __restrict__ in, unsigned short* __restrict__ out, int n) {
  int i = blockIdx.x * 256 + threadIdx.x;
  if (i < n) out[i] = f2bf(in[i]);
}

__global__ __launch_bounds__(256)
void embed_kernel(const int* __restrict__ enc_in, const float* __restrict__ emb,
                  unsigned short* __restrict__ Ae) {
  int idx = blockIdx.x * 256 + threadIdx.x;       // TE*B*E
  int row = idx >> 8, e = idx & 255;
  int tok = enc_in[row];
  Ae[idx] = f2bf(emb[(size_t)tok * Ee + e]);
}

__global__ __launch_bounds__(256)
void pd_kernel(const float* __restrict__ dec_in, const float* __restrict__ Wd,
               const float* __restrict__ bd, unsigned short* __restrict__ pdbf) {
  int idx = blockIdx.x * 256 + threadIdx.x;       // TD*B*H
  int row = idx >> 9, h = idx & (Hh - 1);
  const float* x = dec_in + (size_t)row * 3;
  float v = bd[h] + x[0]*Wd[h] + x[1]*Wd[Hh + h] + x[2]*Wd[2*Hh + h];
  pdbf[idx] = f2bf(v);
}

// ---------------- MDN output head ----------------
__global__ __launch_bounds__(256)
void out_head_kernel(const float* __restrict__ O, const float* __restrict__ bout,
                     float* __restrict__ dout) {
  int row = blockIdx.x * 256 + threadIdx.x;       // TD*B rows
  if (row >= TDd*Bb) return;
  const float* o = O + (size_t)row * 128;
  for (int q = 0; q < 2*NMIXm; ++q)
    dout[O_MUS + (size_t)row * (2*NMIXm) + q] = o[q] + bout[q];
  for (int q = 0; q < 2*NMIXm; ++q)
    dout[O_SIG + (size_t)row * (2*NMIXm) + q] = __expf(o[2*NMIXm + q] + bout[2*NMIXm + q]);
  for (int q = 0; q < NMIXm; ++q)
    dout[O_CORR + (size_t)row * NMIXm + q] = tanhf(o[4*NMIXm + q] + bout[4*NMIXm + q]);
  float mx = -1e30f;
  for (int q = 0; q < NMIXm; ++q) {
    float v = o[5*NMIXm + q] + bout[5*NMIXm + q];
    mx = fmaxf(mx, v);
  }
  float s = 0.0f;
  for (int q = 0; q < NMIXm; ++q)
    s += __expf(o[5*NMIXm + q] + bout[5*NMIXm + q] - mx);
  float ls = logf(s);
  for (int q = 0; q < NMIXm; ++q)
    dout[O_LC + (size_t)row * NMIXm + q] = o[5*NMIXm + q] + bout[5*NMIXm + q] - mx - ls;
  dout[O_BERN + row] = sigm(o[6*NMIXm] + bout[6*NMIXm]);
}

// ---------------- host orchestration ----------------
extern "C" void kernel_launch(void* const* d_in, const int* in_sizes, int n_in,
                              void* d_out, int out_size, void* d_ws, size_t ws_size,
                              hipStream_t stream) {
  (void)in_sizes; (void)n_in; (void)out_size; (void)ws_size;
  const int*   enc_in  = (const int*)  d_in[0];
  const float* dec_in  = (const float*)d_in[1];
  const float* ah_i    = (const float*)d_in[2];
  const float* ac_i    = (const float*)d_in[3];
  const float* ak_i    = (const float*)d_in[4];
  const float* aw_i    = (const float*)d_in[5];
  const float* h1_i    = (const float*)d_in[6];
  const float* c1_i    = (const float*)d_in[7];
  const float* h2_i    = (const float*)d_in[8];
  const float* c2_i    = (const float*)d_in[9];
  const float* encmask = (const float*)d_in[10];
  const float* decmask = (const float*)d_in[11];
  const float* emb     = (const float*)d_in[12];
  const float* We      = (const float*)d_in[13];
  const float* be      = (const float*)d_in[14];
  const float* Wd      = (const float*)d_in[15];
  const float* bd      = (const float*)d_in[16];
  const float* Wax     = (const float*)d_in[17];
  const float* Wah     = (const float*)d_in[18];
  const float* ba      = (const float*)d_in[19];
  const float* Watt    = (const float*)d_in[20];
  const float* batt    = (const float*)d_in[21];
  const float* W1x     = (const float*)d_in[22];
  const float* W1h     = (const float*)d_in[23];
  const float* b1      = (const float*)d_in[24];
  const float* W2x     = (const float*)d_in[25];
  const float* W2h     = (const float*)d_in[26];
  const float* b2      = (const float*)d_in[27];
  const float* Wout    = (const float*)d_in[28];
  const float* bout    = (const float*)d_in[29];
  float* dout = (float*)d_out;

  // workspace carve
  char* ws = (char*)d_ws; size_t off = 0;
  auto carve = [&](size_t bytes) -> void* {
    void* p = ws + off; off += (bytes + 255) & ~(size_t)255; return p;
  };
  typedef unsigned short u16;
  u16* Waxt  = (u16*)carve((size_t)2048*1024*2);
  u16* Waht  = (u16*)carve((size_t)2048*512*2);
  u16* Wattt = (u16*)carve((size_t)32*512*2);
  u16* W1xt  = (u16*)carve((size_t)2048*1536*2);
  u16* W1ht  = (u16*)carve((size_t)2048*512*2);
  u16* W2xt  = (u16*)carve((size_t)2048*1536*2);
  u16* W2ht  = (u16*)carve((size_t)2048*512*2);
  u16* Woutt = (u16*)carve((size_t)128*512*2);
  u16* Wet   = (u16*)carve((size_t)512*256*2);
  u16* Ae    = (u16*)carve((size_t)TEe*Bb*Ee*2);
  float* pe  = (float*)carve((size_t)TEe*Bb*Hh*4);
  u16* pdbf  = (u16*)carve((size_t)TDd*Bb*Hh*2);
  float* Gpre= (float*)carve((size_t)TDd*Bb*4*Hh*4);         // reused for all 3 scans
  u16* hAbf  = (u16*)carve((size_t)(TDd+1)*Bb*Hh*2);         // slot0 = init
  u16* wbf   = (u16*)carve((size_t)(TDd+1)*Bb*Hh*2);
  u16* h1bf  = (u16*)carve((size_t)(TDd+1)*Bb*Hh*2);
  u16* h2bf  = (u16*)carve((size_t)(TDd+1)*Bb*Hh*2);
  float* Obuf= (float*)carve((size_t)TDd*Bb*128*4);

  const size_t BH = (size_t)Bb*Hh;
  dim3 blk128(128), blk256(256);

  // --- weight transposes (f32 (K,N) -> bf16 (Npad,K)) ---
  auto tgrid = [](int Npad, int K){ return dim3((Npad*K + 255)/256); };
  transpose_w_kernel<<<tgrid(2048,1024), blk256, 0, stream>>>(Wax,  Waxt, 1024, 2048, 2048);
  transpose_w_kernel<<<tgrid(2048, 512), blk256, 0, stream>>>(Wah,  Waht,  512, 2048, 2048);
  transpose_w_kernel<<<tgrid(  32, 512), blk256, 0, stream>>>(Watt, Wattt, 512,   30,   32);
  transpose_w_kernel<<<tgrid(2048,1536), blk256, 0, stream>>>(W1x,  W1xt, 1536, 2048, 2048);
  transpose_w_kernel<<<tgrid(2048, 512), blk256, 0, stream>>>(W1h,  W1ht,  512, 2048, 2048);
  transpose_w_kernel<<<tgrid(2048,1536), blk256, 0, stream>>>(W2x,  W2xt, 1536, 2048, 2048);
  transpose_w_kernel<<<tgrid(2048, 512), blk256, 0, stream>>>(W2h,  W2ht,  512, 2048, 2048);
  transpose_w_kernel<<<tgrid( 128, 512), blk256, 0, stream>>>(Wout, Woutt, 512,  121,  128);
  transpose_w_kernel<<<tgrid( 512, 256), blk256, 0, stream>>>(We,   Wet,   256,  512,  512);

  // --- initial states to bf16 (slot 0 of each sequence buffer) ---
  cvt_bf16_kernel<<<dim3(128), blk256, 0, stream>>>(ah_i, hAbf, (int)BH);
  cvt_bf16_kernel<<<dim3(128), blk256, 0, stream>>>(aw_i, wbf,  (int)BH);
  cvt_bf16_kernel<<<dim3(128), blk256, 0, stream>>>(h1_i, h1bf, (int)BH);
  cvt_bf16_kernel<<<dim3(128), blk256, 0, stream>>>(h2_i, h2bf, (int)BH);

  // --- p_e = emb[enc_in] @ We + be  (bf16 WMMA GEMM) ---
  embed_kernel<<<dim3((TEe*Bb*Ee)/256), blk256, 0, stream>>>(enc_in, emb, Ae);
  gemm_bf16<<<dim3(Hh/64, (TEe*Bb)/128), blk128, 0, stream>>>(Ae, Ee, Wet, Ee, pe, Hh, Ee, be, 0);

  // --- p_d (K=3, VALU) directly to bf16 ---
  pd_kernel<<<dim3((TDd*Bb*Hh)/256), blk256, 0, stream>>>(dec_in, Wd, bd, pdbf);

  // --- attention scan: Gpre = p_d @ Wax[:H] + ba, then 512 fused steps ---
  gemm_bf16<<<dim3(4*Hh/64, (TDd*Bb)/128), blk128, 0, stream>>>(pdbf, Hh, Waxt, 1024, Gpre, 4*Hh, Hh, ba, 0);
  for (int t = 0; t < TDd; ++t) {
    const float* hp = t ? (dout + O_AH + (size_t)(t-1)*BH) : ah_i;
    const float* cp = t ? (dout + O_AC + (size_t)(t-1)*BH) : ac_i;
    const float* kp = t ? (dout + O_AK + (size_t)(t-1)*Bb*KATT) : ak_i;
    const float* wp = t ? (dout + O_AW + (size_t)(t-1)*BH) : aw_i;
    const float* mt = decmask + (size_t)t*Bb;
    lstm_step<<<dim3(Hh/16), blk128, 0, stream>>>(
        Gpre + (size_t)t*Bb*4*Hh,
        wbf + (size_t)t*BH, Waxt + 512, 1024,      // w part (K-offset 512 of Wax)
        hAbf + (size_t)t*BH, Waht, 512,            // h part
        hp, cp, mt,
        dout + O_AH + (size_t)t*BH, dout + O_AC + (size_t)t*BH,
        hAbf + (size_t)(t+1)*BH);
    attn_phi_kernel<<<dim3(1), blk256, 0, stream>>>(
        hAbf + (size_t)(t+1)*BH, Wattt, batt, kp, mt, encmask,
        dout + O_AK + (size_t)t*Bb*KATT, dout + O_APHI + (size_t)t*Bb*TEe);
    attn_w_kernel<<<dim3(BH/256), blk256, 0, stream>>>(
        dout + O_APHI + (size_t)t*Bb*TEe, pe, wp, mt,
        dout + O_AW + (size_t)t*BH, wbf + (size_t)(t+1)*BH);
  }

  // --- layer 1: Gpre = [p_d, attn_w, attn_h] @ W1x + b1, then scan ---
  gemm_bf16<<<dim3(4*Hh/64, (TDd*Bb)/128), blk128, 0, stream>>>(pdbf,      Hh, W1xt,        1536, Gpre, 4*Hh, Hh, b1,   0);
  gemm_bf16<<<dim3(4*Hh/64, (TDd*Bb)/128), blk128, 0, stream>>>(wbf + BH,  Hh, W1xt +  512, 1536, Gpre, 4*Hh, Hh, NULL, 1);
  gemm_bf16<<<dim3(4*Hh/64, (TDd*Bb)/128), blk128, 0, stream>>>(hAbf + BH, Hh, W1xt + 1024, 1536, Gpre, 4*Hh, Hh, NULL, 1);
  for (int t = 0; t < TDd; ++t) {
    const float* hp = t ? (dout + O_H1 + (size_t)(t-1)*BH) : h1_i;
    const float* cp = t ? (dout + O_C1 + (size_t)(t-1)*BH) : c1_i;
    lstm_step<<<dim3(Hh/16), blk128, 0, stream>>>(
        Gpre + (size_t)t*Bb*4*Hh,
        h1bf + (size_t)t*BH, W1ht, 512,
        NULL, NULL, 0,
        hp, cp, decmask + (size_t)t*Bb,
        dout + O_H1 + (size_t)t*BH, dout + O_C1 + (size_t)t*BH,
        h1bf + (size_t)(t+1)*BH);
  }

  // --- layer 2: Gpre = [p_d, attn_w, h1] @ W2x + b2, then scan ---
  gemm_bf16<<<dim3(4*Hh/64, (TDd*Bb)/128), blk128, 0, stream>>>(pdbf,      Hh, W2xt,        1536, Gpre, 4*Hh, Hh, b2,   0);
  gemm_bf16<<<dim3(4*Hh/64, (TDd*Bb)/128), blk128, 0, stream>>>(wbf + BH,  Hh, W2xt +  512, 1536, Gpre, 4*Hh, Hh, NULL, 1);
  gemm_bf16<<<dim3(4*Hh/64, (TDd*Bb)/128), blk128, 0, stream>>>(h1bf + BH, Hh, W2xt + 1024, 1536, Gpre, 4*Hh, Hh, NULL, 1);
  for (int t = 0; t < TDd; ++t) {
    const float* hp = t ? (dout + O_H2 + (size_t)(t-1)*BH) : h2_i;
    const float* cp = t ? (dout + O_C2 + (size_t)(t-1)*BH) : c2_i;
    lstm_step<<<dim3(Hh/16), blk128, 0, stream>>>(
        Gpre + (size_t)t*Bb*4*Hh,
        h2bf + (size_t)t*BH, W2ht, 512,
        NULL, NULL, 0,
        hp, cp, decmask + (size_t)t*Bb,
        dout + O_H2 + (size_t)t*BH, dout + O_C2 + (size_t)t*BH,
        h2bf + (size_t)(t+1)*BH);
  }

  // --- output head: O = h2 @ Wout (padded N=128), then MDN transforms ---
  gemm_bf16<<<dim3(128/64, (TDd*Bb)/128), blk128, 0, stream>>>(h2bf + BH, Hh, Woutt, Hh, Obuf, 128, Hh, NULL, 0);
  out_head_kernel<<<dim3((TDd*Bb)/256), blk256, 0, stream>>>(Obuf, bout, dout);
}
